// MaceNet_40647570489450
// MI455X (gfx1250) — matrix operand order repaired and checked by reading, not compiled, not run on hardware
//
#include <hip/hip_runtime.h>
#include <hip/hip_bf16.h>

typedef __attribute__((ext_vector_type(16))) _Float16 v16h;
typedef __attribute__((ext_vector_type(8)))  float    v8f;
typedef __attribute__((ext_vector_type(4)))  unsigned v4u;

union Frag16 { v16h v; v4u q[2]; };

constexpr int   NNODE = 768;
constexpr int   FDIM  = 64;
constexpr int   VDIM  = 16;
constexpr int   NBES  = 8;
constexpr int   HRAD  = 64;
constexpr int   TSTEP = 2;
constexpr int   PDIM  = 176;                 // F + V + F + V + V
constexpr float RMAXF = 10.0f;
constexpr float INV_AVG = 1.0f / 767.0f;

// ---- workspace layout (float offsets) ----
constexpr int OFF_HS_STATE = 0;                                  // N*F
constexpr int OFF_HV_STATE = OFF_HS_STATE + NNODE * FDIM;        // N*V*3
constexpr int OFF_HS_EDGE  = OFF_HV_STATE + NNODE * VDIM * 3;    // N*F   (hs = h_s @ w_node_s)
constexpr int OFF_HV_EDGE  = OFF_HS_EDGE  + NNODE * FDIM;        // N*V*3 (hv = mixed h_v)
constexpr int OFF_AGG_S    = OFF_HV_EDGE  + NNODE * VDIM * 3;    // N*80
constexpr int OFF_AGG_V    = OFF_AGG_S    + NNODE * 80;          // N*96*3
constexpr int OFF_COM      = OFF_AGG_V    + NNODE * 96 * 3;      // 3
constexpr int OFF_W2T      = ((OFF_COM + 3 + 3) / 4) * 4;        // 16B aligned; T*P*HR halves

__device__ __forceinline__ float fast_rcp(float x)  { return __builtin_amdgcn_rcpf(x); }
__device__ __forceinline__ float fast_silu(float a) { return a * __builtin_amdgcn_rcpf(1.f + __expf(-a)); }
__device__ __forceinline__ float fast_sig(float a)  { return __builtin_amdgcn_rcpf(1.f + __expf(-a)); }

// gfx1250 async global->LDS copy (ISA: GLOBAL_LOAD_ASYNC_TO_LDS_B128, ASYNCcnt).
// Per-lane: each lane supplies its own global address and LDS byte offset
// (generic __shared__ pointers carry the LDS offset in their low 32 bits).
__device__ __forceinline__ void async_copy16(void* lds_dst, const void* gsrc) {
  asm volatile("global_load_async_to_lds_b128 %0, %1, off"
               :: "v"((unsigned)(unsigned long long)lds_dst), "v"(gsrc)
               : "memory");
}
__device__ __forceinline__ void wait_async0() {
  asm volatile("s_wait_asynccnt 0x0" ::: "memory");
}

// ---------------------------------------------------------------------------
// One-time per launch: transpose w_rad2 -> f16 [t][p][k] (B-fragment friendly),
// and compute center of mass.
// ---------------------------------------------------------------------------
__global__ void k_prep(const float* __restrict__ x, const float* __restrict__ w_rad2,
                       _Float16* __restrict__ w2t, float* __restrict__ com) {
  const int idx = blockIdx.x * blockDim.x + threadIdx.x;
  if (idx < TSTEP * PDIM * HRAD) {
    const int t = idx / (PDIM * HRAD);
    const int r = idx % (PDIM * HRAD);
    const int p = r / HRAD;
    const int k = r % HRAD;
    w2t[idx] = (_Float16)w_rad2[(t * HRAD + k) * PDIM + p];
  }
  if (idx < 3) {
    float s = 0.f;
    for (int n = 0; n < NNODE; ++n) s += x[n * 3 + idx];
    com[idx] = s / (float)NNODE;
  }
}

// ---------------------------------------------------------------------------
// Init node state and zero the output accumulators.
// ---------------------------------------------------------------------------
__global__ void k_init(const float* __restrict__ embed, float* __restrict__ h_s,
                       float* __restrict__ h_v, float* __restrict__ out) {
  const int idx = blockIdx.x * blockDim.x + threadIdx.x;
  if (idx < NNODE * FDIM)     h_s[idx] = embed[idx & (FDIM - 1)];
  if (idx < NNODE * VDIM * 3) h_v[idx] = 0.f;
  if (idx < NNODE * (32 + 48)) out[idx] = 0.f;
}

// ---------------------------------------------------------------------------
// Per-step node pre-transform: hs = h_s @ w_node_s[t];  hv = einsum(h_v, w_node_v[t])
// ---------------------------------------------------------------------------
__global__ void k_node_pre(const float* __restrict__ h_s, const float* __restrict__ h_v,
                           const float* __restrict__ w_node_s, const float* __restrict__ w_node_v,
                           float* __restrict__ hs, float* __restrict__ hv, int t) {
  const int idx = blockIdx.x * blockDim.x + threadIdx.x;
  if (idx < NNODE * FDIM) {
    const int n = idx >> 6, f = idx & 63;
    float a = 0.f;
    for (int k = 0; k < FDIM; ++k) a += h_s[n * FDIM + k] * w_node_s[(t * FDIM + k) * FDIM + f];
    hs[idx] = a;
  } else if (idx < NNODE * FDIM + NNODE * VDIM * 3) {
    const int r = idx - NNODE * FDIM;
    const int n = r / 48, rr = r % 48, w = rr / 3, c = rr % 3;
    float a = 0.f;
    for (int v = 0; v < VDIM; ++v) a += h_v[n * 48 + v * 3 + c] * w_node_v[(t * VDIM + v) * VDIM + w];
    hv[r] = a;
  }
}

// ---------------------------------------------------------------------------
// Fused edge kernel. One block per receiver i; 4 waves; 48 sender tiles of 16.
// Async global->LDS staging, double-buffered -> ONE barrier per tile.
// Geometry + bessel (registers, Chebyshev) -> VALU layer1 -> WMMA f16 layer2
// (11 N-tiles) -> envelope + messages -> register aggregation.
// ---------------------------------------------------------------------------
__launch_bounds__(128, 1)
__global__ void k_edge(const float* __restrict__ x,
                       const float* __restrict__ hs,
                       const float* __restrict__ hv,
                       const float* __restrict__ w_rad1,
                       const float* __restrict__ b_rad1,
                       const _Float16* __restrict__ w2t,
                       float* __restrict__ agg_s,
                       float* __restrict__ agg_v,
                       int t) {
  const int i   = blockIdx.x;
  const int tid = threadIdx.x;
  // wave id as an SGPR so kind-dispatch branches are scalar, not exec-mask churn
  const int wvi = __builtin_amdgcn_readfirstlane(tid >> 5);
  const int ln  = tid & 31;        // lane 0..31
  const int l16 = ln & 15;
  const bool hi = ln >= 16;
  const int eb  = hi ? 8 : 0;      // epilogue edge base for this half-wave

  __shared__ __align__(16) float sh_hs[2][16 * FDIM];     // sender scalar feats
  __shared__ __align__(16) float sh_ev[2][16 * VDIM * 3]; // sender vector feats
  __shared__ float sh_rhat[2][16 * 3];
  __shared__ float sh_env[2][16];
  __shared__ float sh_W1[NBES * HRAD];
  __shared__ float sh_b1[HRAD];
  __shared__ __align__(16) _Float16 sh_hidden[2][16 * HRAD];

  for (int idx = tid; idx < NBES * HRAD; idx += 128) sh_W1[idx] = w_rad1[t * NBES * HRAD + idx];
  if (tid < HRAD) sh_b1[tid] = b_rad1[t * HRAD + tid];

  // wave -> N-tiles {wvi, wvi+4, wvi+8}; B fragments resident in VGPRs all block.
  const int nq = (wvi < 3) ? 3 : 2;
  Frag16 B[3][2];
#pragma unroll
  for (int q = 0; q < 3; ++q) {
    if (q < nq) {
      const int ncol = (wvi + 4 * q) * 16 + l16;
      const _Float16* bp = w2t + ((size_t)(t * PDIM + ncol)) * HRAD + (hi ? 8 : 0);
#pragma unroll
      for (int kc = 0; kc < 2; ++kc) {
        B[q][kc].q[0] = *(const v4u*)(bp + kc * 32);
        B[q][kc].q[1] = *(const v4u*)(bp + kc * 32 + 16);
      }
    }
  }

  float acc[3][3];
#pragma unroll
  for (int q = 0; q < 3; ++q)
#pragma unroll
    for (int c = 0; c < 3; ++c) acc[q][c] = 0.f;

  const float xi0 = x[i * 3 + 0], xi1 = x[i * 3 + 1], xi2 = x[i * 3 + 2];

  __syncthreads();                           // publish W1/b1 once

  for (int tile = 0; tile < NNODE / 16; ++tile) {
    const int b  = tile & 1;                 // double buffer: no loop-top barrier
    const int j0 = tile * 16;

    // async staging into LDS buffer b (completes while we do geometry+layer1):
    // hs tile = 256 x 16B, ev tile = 192 x 16B, per-lane addresses.
    {
      char*       dh = (char*)sh_hs[b];
      const char* sh = (const char*)(hs + j0 * FDIM);
      async_copy16(dh + tid * 16,          sh + tid * 16);
      async_copy16(dh + (tid + 128) * 16,  sh + (tid + 128) * 16);
      char*       dv = (char*)sh_ev[b];
      const char* sv = (const char*)(hv + j0 * VDIM * 3);
      async_copy16(dv + tid * 16, sv + tid * 16);
      if (tid < 64) async_copy16(dv + (tid + 128) * 16, sv + (tid + 128) * 16);
    }
    if (tile + 2 < NNODE / 16)
      __builtin_prefetch(hs + (j0 + 32) * FDIM + tid * 8, 0, 0);

    // every lane computes the geometry of "its" edge e = l16 into registers
    float rbv[NBES];
    {
      const int j = j0 + l16;
      const float dx = xi0 - x[j * 3 + 0];
      const float dy = xi1 - x[j * 3 + 1];
      const float dz = xi2 - x[j * 3 + 2];
      const bool self = (j == i);
      const float r = self ? 1.f : sqrtf(dx * dx + dy * dy + dz * dz);
      const float inv = fast_rcp(r);
      const float c0i = 0.44721359549995794f * inv;       // sqrt(2/RMAX)/r
      const float w0  = 3.14159265358979f / RMAXF;
      // sin((n+1)th) via Chebyshev recurrence: 1 sincos instead of 8 sins
      float s1, c1;
      __sincosf(w0 * r, &s1, &c1);
      const float twoc = 2.f * c1;
      float sA = s1;                    // sin(1*th)
      float sB = twoc * s1 * c1 - 0.f;  // placeholder value replaced below
      sB = 2.f * s1 * c1;               // sin(2*th)
      rbv[0] = c0i * sA;
      rbv[1] = c0i * sB;
#pragma unroll
      for (int nb = 2; nb < NBES; ++nb) {
        const float sC = twoc * sB - sA;
        rbv[nb] = c0i * sC;
        sA = sB; sB = sC;
      }
      if (wvi == 0 && !hi) {                 // one copy published for the epilogue
        sh_rhat[b][l16 * 3 + 0] = dx * inv;
        sh_rhat[b][l16 * 3 + 1] = dy * inv;
        sh_rhat[b][l16 * 3 + 2] = dz * inv;
        const float u = 2.f * (1.f - r * (1.f / RMAXF));
        sh_env[b][l16] = (!self && u > 0.f) ? 1.2f * __expf(-fast_rcp(u)) : 0.f;
      }
    }

    // layer 1: hidden[e][h] = silu(rb[e] @ W1 + b1) -> f16 tile, row-major
    {
      const int e  = l16;
      const int hb = wvi * 16 + eb;
#pragma unroll
      for (int hh = 0; hh < 8; ++hh) {
        const int h = hb + hh;
        float a = sh_b1[h];
#pragma unroll
        for (int k = 0; k < NBES; ++k) a += rbv[k] * sh_W1[k * HRAD + h];
        sh_hidden[b][e * HRAD + h] = (_Float16)fast_silu(a);
      }
    }
    wait_async0();                           // my staging copies have landed in LDS
    __syncthreads();                         // the ONLY barrier per tile

    // A fragments (ISA 16-bit A 16x32 layout): lane m=l16, kb=(hi?8:0)
    Frag16 A0, A1;
    {
      const _Float16* ap = sh_hidden[b] + l16 * HRAD + eb;
      A0.q[0] = *(const v4u*)(ap);
      A0.q[1] = *(const v4u*)(ap + 16);
      A1.q[0] = *(const v4u*)(ap + 32);
      A1.q[1] = *(const v4u*)(ap + 48);
    }

    const float* shs = sh_hs[b];
    const float* sev = sh_ev[b];
    const float* srh = sh_rhat[b];
    const float* sen = sh_env[b];

#pragma unroll
    for (int q = 0; q < 3; ++q) {
      if (q >= nq) continue;
      const int nt = wvi + 4 * q;            // scalar (wvi is readfirstlane'd)
      v8f d = {};
      d = __builtin_amdgcn_wmma_f32_16x16x32_f16(false, A0.v, false, B[q][0].v, (short)0, d, false, false);
      d = __builtin_amdgcn_wmma_f32_16x16x32_f16(false, A1.v, false, B[q][1].v, (short)0, d, false, false);
      // C layout: element p is edge (p + 8*half), column l16 of this N-tile.
      float t0 = 0.f, t1 = 0.f, t2 = 0.f;
      if (nt < 4) {                                     // msg_s[:64] = es*r1
#pragma unroll
        for (int p = 0; p < 8; ++p) {
          const int e = p + eb;
          t0 += shs[e * FDIM + nt * 16 + l16] * (d[p] * sen[e]);
        }
      } else if (nt == 4) {                             // msg_s[64:] = (ev.rhat)*r2
#pragma unroll
        for (int p = 0; p < 8; ++p) {
          const int e = p + eb;
          const float* evp = sev + e * VDIM * 3 + l16 * 3;
          const float* rh  = srh + e * 3;
          t0 += (evp[0] * rh[0] + evp[1] * rh[1] + evp[2] * rh[2]) * (d[p] * sen[e]);
        }
      } else if (nt < 9) {                              // v1 = es*r3 (x) rhat
#pragma unroll
        for (int p = 0; p < 8; ++p) {
          const int e = p + eb;
          const float s = shs[e * FDIM + (nt - 5) * 16 + l16] * (d[p] * sen[e]);
          const float* rh = srh + e * 3;
          t0 += s * rh[0]; t1 += s * rh[1]; t2 += s * rh[2];
        }
      } else if (nt == 9) {                             // v2 = ev*r4
#pragma unroll
        for (int p = 0; p < 8; ++p) {
          const int e = p + eb;
          const float Re = d[p] * sen[e];
          const float* evp = sev + e * VDIM * 3 + l16 * 3;
          t0 += evp[0] * Re; t1 += evp[1] * Re; t2 += evp[2] * Re;
        }
      } else {                                          // v3 = cross(ev,rhat)*r5
#pragma unroll
        for (int p = 0; p < 8; ++p) {
          const int e = p + eb;
          const float Re = d[p] * sen[e];
          const float* evp = sev + e * VDIM * 3 + l16 * 3;
          const float* rh  = srh + e * 3;
          t0 += (evp[1] * rh[2] - evp[2] * rh[1]) * Re;
          t1 += (evp[2] * rh[0] - evp[0] * rh[2]) * Re;
          t2 += (evp[0] * rh[1] - evp[1] * rh[0]) * Re;
        }
      }
      acc[q][0] += t0; acc[q][1] += t1; acc[q][2] += t2;
    }
  }

  // combine half-wave partials (edges 0-7 vs 8-15 of each tile), wave32 xor-16
#pragma unroll
  for (int q = 0; q < 3; ++q)
#pragma unroll
    for (int c = 0; c < 3; ++c)
      acc[q][c] += __shfl_xor(acc[q][c], 16, 32);

  if (!hi) {
#pragma unroll
    for (int q = 0; q < 3; ++q) {
      if (q >= nq) continue;
      const int nt = wvi + 4 * q;
      if (nt < 4) {
        agg_s[i * 80 + nt * 16 + l16] = acc[q][0];
      } else if (nt == 4) {
        agg_s[i * 80 + 64 + l16] = acc[q][0];
      } else {
        const int f = (nt < 9) ? ((nt - 5) * 16 + l16) : ((nt == 9) ? (64 + l16) : (80 + l16));
        agg_v[(i * 96 + f) * 3 + 0] = acc[q][0];
        agg_v[(i * 96 + f) * 3 + 1] = acc[q][1];
        agg_v[(i * 96 + f) * 3 + 2] = acc[q][2];
      }
    }
  }
}

// ---------------------------------------------------------------------------
// Per-step node update + readout accumulation. One block per node.
// ---------------------------------------------------------------------------
__launch_bounds__(128, 1)
__global__ void k_node_post(const float* __restrict__ agg_s, const float* __restrict__ agg_v,
                            const float* __restrict__ w_mix_s, const float* __restrict__ w_mix_v,
                            const float* __restrict__ w_gate, const float* __restrict__ w_ro_s,
                            const float* __restrict__ w_ro_v, float* __restrict__ h_s,
                            float* __restrict__ h_v, float* __restrict__ out, int t) {
  const int i = blockIdx.x;
  const int tid = threadIdx.x;
  __shared__ float a_s[80];
  __shared__ float a_v[96 * 3];
  __shared__ float m_s[64];
  __shared__ float hsn[64];
  __shared__ float mv[48];
  __shared__ float gate[16];
  for (int idx = tid; idx < 80; idx += 128)  a_s[idx] = agg_s[i * 80 + idx] * INV_AVG;
  for (int idx = tid; idx < 288; idx += 128) a_v[idx] = agg_v[i * 288 + idx] * INV_AVG;
  __syncthreads();
  if (tid < 64) {
    float a = 0.f;
    for (int p = 0; p < 80; ++p) a += a_s[p] * w_mix_s[(t * 80 + p) * 64 + tid];
    m_s[tid] = a;
    hsn[tid] = fast_silu(a);
  } else if (tid < 112) {
    const int r = tid - 64, v = r / 3, c = r % 3;
    float a = 0.f;
    for (int p = 0; p < 96; ++p) a += a_v[p * 3 + c] * w_mix_v[(t * 96 + p) * 16 + v];
    mv[r] = a;
  }
  __syncthreads();
  if (tid < 16) {
    float a = 0.f;
    for (int f = 0; f < 64; ++f) a += m_s[f] * w_gate[(t * 64 + f) * 16 + tid];
    gate[tid] = fast_sig(a);
  }
  __syncthreads();
  if (tid < 48) mv[tid] *= gate[tid / 3];              // h_v = gate * m_v
  __syncthreads();
  if (tid < 32) {
    float a = 0.f;
    for (int f = 0; f < 64; ++f) a += hsn[f] * w_ro_s[(t * 64 + f) * 32 + tid];
    out[i * 32 + tid] += a;
  } else if (tid >= 64 && tid < 112) {
    const int r = tid - 64, k = r / 3, c = r % 3;
    float a = 0.f;
    for (int v = 0; v < 16; ++v) a += mv[v * 3 + c] * w_ro_v[(t * 16 + v) * 16 + k];
    out[NNODE * 32 + i * 48 + r] += a;
  }
  if (tid < 64)       h_s[i * 64 + tid] = hsn[tid];
  else if (tid < 112) h_v[i * 48 + (tid - 64)] = mv[tid - 64];
}

__global__ void k_final(const float* __restrict__ com, float* __restrict__ out) {
  const int idx = blockIdx.x * blockDim.x + threadIdx.x;
  if (idx < NNODE * 48) out[NNODE * 32 + idx] += com[idx % 3];
}

// ---------------------------------------------------------------------------
extern "C" void kernel_launch(void* const* d_in, const int* in_sizes, int n_in,
                              void* d_out, int out_size, void* d_ws, size_t ws_size,
                              hipStream_t stream) {
  (void)in_sizes; (void)n_in; (void)out_size; (void)ws_size;
  const float* x        = (const float*)d_in[0];
  const float* embed    = (const float*)d_in[1];
  const float* w_node_s = (const float*)d_in[2];
  const float* w_node_v = (const float*)d_in[3];
  const float* w_rad1   = (const float*)d_in[4];
  const float* b_rad1   = (const float*)d_in[5];
  const float* w_rad2   = (const float*)d_in[6];
  const float* w_mix_s  = (const float*)d_in[7];
  const float* w_mix_v  = (const float*)d_in[8];
  const float* w_gate   = (const float*)d_in[9];
  const float* w_ro_s   = (const float*)d_in[10];
  const float* w_ro_v   = (const float*)d_in[11];
  float* out = (float*)d_out;
  float* ws  = (float*)d_ws;

  float*    h_s   = ws + OFF_HS_STATE;
  float*    h_v   = ws + OFF_HV_STATE;
  float*    hs_e  = ws + OFF_HS_EDGE;
  float*    hv_e  = ws + OFF_HV_EDGE;
  float*    agg_s = ws + OFF_AGG_S;
  float*    agg_v = ws + OFF_AGG_V;
  float*    com   = ws + OFF_COM;
  _Float16* w2t   = (_Float16*)(ws + OFF_W2T);

  k_prep<<<(TSTEP * PDIM * HRAD + 255) / 256, 256, 0, stream>>>(x, w_rad2, w2t, com);
  k_init<<<(NNODE * 80 + 255) / 256, 256, 0, stream>>>(embed, h_s, h_v, out);
  for (int t = 0; t < TSTEP; ++t) {
    k_node_pre<<<(NNODE * (FDIM + VDIM * 3) + 255) / 256, 256, 0, stream>>>(
        h_s, h_v, w_node_s, w_node_v, hs_e, hv_e, t);
    k_edge<<<NNODE, 128, 0, stream>>>(x, hs_e, hv_e, w_rad1, b_rad1, w2t, agg_s, agg_v, t);
    k_node_post<<<NNODE, 128, 0, stream>>>(agg_s, agg_v, w_mix_s, w_mix_v, w_gate,
                                           w_ro_s, w_ro_v, h_s, h_v, out, t);
  }
  k_final<<<(NNODE * 48 + 255) / 256, 256, 0, stream>>>(com, out);
}
